// DeltaRuleMemory_62560493633832
// MI455X (gfx1250) — matrix operand order repaired and checked by reading, not compiled, not run on hardware
//
#include <hip/hip_runtime.h>
#include <hip/hip_bf16.h>
#include <stdint.h>

// ---------------------------------------------------------------------------
// DeltaRuleMemory on gfx1250:
//   B=8, S=4096, DK=DV=256.  M rows are independent -> 16 workgroups each own
//   a 16x256 slice of M as persistent V_WMMA f32 accumulators.  bf16 WMMA for
//   all three per-step GEMMs, async global->LDS double-buffered staging,
//   ds_add_f32 cross-wave K-reduction.  err phase is branchless (lane mask)
//   to keep the serial per-step chain free of exec-mask gymnastics.
// ---------------------------------------------------------------------------

typedef __bf16 bf16_t;
typedef __attribute__((ext_vector_type(16))) __bf16 v16bf;
typedef __attribute__((ext_vector_type(8)))  __bf16 v8bf;
typedef __attribute__((ext_vector_type(8)))  float  v8f;
typedef __attribute__((ext_vector_type(4)))  float  v4f;

#define DR_B 8
#define DR_S 4096
#define DR_D 256
#define LR_OVER_B (0.1f / 8.0f)

struct __align__(64) Smem {
  float  sK[2][DR_B * DR_D];   // async staging, double buffered (8 KB each)
  float  sV[2][DR_B * DR_D];
  float  sQ[2][DR_B * DR_D];
  bf16_t Mbf[16][DR_D];        // bf16 mirror of this WG's M slice (A-operand)
  bf16_t kT[DR_D][16];         // k transposed, dk-major (update B-operand)
  bf16_t kB[DR_B][DR_D];       // k b-major (y_pred B-operand)
  bf16_t qB[DR_B][DR_D];       // q b-major (y B-operand)
  bf16_t errT[16][32];         // scaled err^T, K-padded to 32 (update A-op)
  bf16_t zpad[16];             // 32B of zeros for padded B-operand lanes
  float  redP[16][16];         // y_pred^T cross-wave reduction tile
  float  redY[16][16];         // y^T cross-wave reduction tile
};

__device__ inline uint32_t lds_off(const void* p) {
  return (uint32_t)(uintptr_t)p;   // flat shared addr low 32 bits == LDS offset
}

// one 16-byte async global->LDS copy (tracked by ASYNCcnt)
__device__ inline void async_cp16(const void* lds, const float* g) {
  asm volatile("global_load_async_to_lds_b128 %0, %1, off"
               :: "v"(lds_off(lds)), "v"(g) : "memory");
}

__device__ inline v16bf ld_cont32(const bf16_t* p) {   // 16 contiguous bf16
  v16bf r;
  ((v8bf*)&r)[0] = *(const v8bf*)(p);
  ((v8bf*)&r)[1] = *(const v8bf*)(p + 8);
  return r;
}
__device__ inline v16bf ld_split(const bf16_t* p) {    // 8 @ p, 8 @ p+16
  v16bf r;
  ((v8bf*)&r)[0] = *(const v8bf*)(p);
  ((v8bf*)&r)[1] = *(const v8bf*)(p + 16);
  return r;
}

template <bool RA>
__device__ inline v8f wmma_bf16(v16bf a, v16bf b, v8f c) {
  return __builtin_amdgcn_wmma_f32_16x16x32_bf16(
      /*neg_a=*/false, a, /*neg_b=*/false, b,
      /*c_mod=*/(short)0, c, /*reuse_a=*/RA, /*reuse_b=*/false);
}

// stage step t's k/v/q into LDS buffer `buf` : 6 async b128 ops per lane
__device__ inline void stage(Smem& s, const float* K, const float* V,
                             const float* Q, int t, int buf, int tid) {
#pragma unroll
  for (int c = 0; c < 2; ++c) {
    int f  = (c * 256 + tid) * 4;          // flat f32 index 0..2047
    int b  = f >> 8;
    int dk = f & (DR_D - 1);
    size_t go = ((size_t)b * DR_S + (size_t)t) * DR_D + (size_t)dk;
    async_cp16(&s.sK[buf][f], K + go);
    async_cp16(&s.sV[buf][f], V + go);
    async_cp16(&s.sQ[buf][f], Q + go);
  }
}

__global__ __launch_bounds__(256, 1)
void delta_rule_scan(const float* __restrict__ K, const float* __restrict__ V,
                     const float* __restrict__ Q, const float* __restrict__ M0,
                     float* __restrict__ Y) {
  __shared__ Smem s;
  const int tid  = threadIdx.x;
  const int lane = tid & 31;
  const int wave = tid >> 5;     // 8 waves
  const int l16  = lane & 15;
  const int half = lane >> 4;
  const int dvb  = blockIdx.x * 16;   // this WG's global DV base

  // ---------------- one-time LDS init (zero pads / reduction tiles) --------
  for (int i = tid; i < 16 * 32; i += 256) ((bf16_t*)s.errT)[i] = (bf16_t)0.0f;
  if (tid < 16) s.zpad[tid] = (bf16_t)0.0f;
  s.redP[tid >> 4][tid & 15] = 0.0f;
  s.redY[tid >> 4][tid & 15] = 0.0f;
  for (int i = tid; i < DR_D; i += 256)        // kT pad columns b=8..15
#pragma unroll
    for (int b = 8; b < 16; ++b) s.kT[i][b] = (bf16_t)0.0f;

  // persistent M accumulators: wave owns dk tiles u = 2w, 2w+1 (16 cols each)
  v8f macc[2];
#pragma unroll
  for (int u2 = 0; u2 < 2; ++u2) {
    const int u = 2 * wave + u2;
#pragma unroll
    for (int r = 0; r < 8; ++r) {
      const int m  = half * 8 + r;
      const int dk = u * 16 + l16;
      const float v0 = M0[(size_t)(dvb + m) * DR_D + dk];
      macc[u2][r] = v0;
      s.Mbf[m][dk] = (bf16_t)v0;
    }
  }

  // lane mask / clamped b-index for the branchless err phase
  const float bmask = (l16 < DR_B) ? (LR_OVER_B) : 0.0f;
  const int   bclmp = l16 & (DR_B - 1);

  stage(s, K, V, Q, 0, 0, tid);    // prologue: stage t=0

#pragma unroll 1
  for (int t = 0; t < DR_S; ++t) {
    const int buf = t & 1;
    const int tn  = (t + 1 < DR_S) ? (t + 1) : (DR_S - 1);
    stage(s, K, V, Q, tn, buf ^ 1, tid);            // prefetch next step
    asm volatile("s_wait_asynccnt 0x6" ::: "memory"); // step t's 6 done
    __syncthreads();                                 // (A) staged visible

    // -------- convert staged f32 -> bf16 operand layouts -------------------
    {
      const int dk = tid;                            // kT: dk-major
      bf16_t tt[8];
#pragma unroll
      for (int b = 0; b < 8; ++b) tt[b] = (bf16_t)s.sK[buf][b * DR_D + dk];
      *(v8bf*)&s.kT[dk][0] = *(v8bf*)tt;

      const int b2 = tid >> 5, d0 = (tid & 31) * 8;  // kB/qB: b-major
      bf16_t tk[8], tq[8];
#pragma unroll
      for (int i = 0; i < 8; ++i) {
        tk[i] = (bf16_t)s.sK[buf][b2 * DR_D + d0 + i];
        tq[i] = (bf16_t)s.sQ[buf][b2 * DR_D + d0 + i];
      }
      *(v8bf*)&s.kB[b2][d0] = *(v8bf*)tk;
      *(v8bf*)&s.qB[b2][d0] = *(v8bf*)tq;
    }
    __syncthreads();                                 // (B)

    // -------- y_pred^T partial: A = M slice (bf16 mirror), B = k, K=32*wave
    {
      const int g = wave;                            // K-group 32*g..32*g+31
      v16bf a = ld_cont32(&s.Mbf[l16][g * 32 + half * 16]);
      const bf16_t* bp = (l16 < DR_B) ? &s.kB[l16][g * 32 + half * 16]
                                      : s.zpad;
      v16bf bm = ld_cont32(bp);
      v8f zc = {0.f, 0.f, 0.f, 0.f, 0.f, 0.f, 0.f, 0.f};
      v8f acc = wmma_bf16<false>(a, bm, zc);
#pragma unroll
      for (int r = 0; r < 8; ++r)
        atomicAdd(&s.redP[half * 8 + r][l16], acc[r]);
    }
    __syncthreads();                                 // (C)

    // -------- err^T = scale*(v^T - y_pred^T), bf16, wave 0, branchless -----
    if (wave == 0) {
#pragma unroll
      for (int r = 0; r < 8; ++r) {
        const int m = half * 8 + r;
        const float yp = s.redP[m][l16];
        const float vv = s.sV[buf][bclmp * DR_D + dvb + m];
        const float e  = (vv - yp) * bmask;          // 0 for padded lanes
        s.errT[m][l16] = (bf16_t)e;
        s.redP[m][l16] = 0.0f;                       // re-zero for next step
      }
    }
    __syncthreads();                                 // (D)

    // -------- rank-8 update: M tiles += err^T x k (accumulate in regs) -----
    {
      v16bf a = ld_cont32(&s.errT[l16][half * 16]);
      const int u0 = 2 * wave;
      const bf16_t* bp0 = (half == 0) ? &s.kT[u0 * 16 + l16][0] : s.zpad;
      const bf16_t* bp1 = (half == 0) ? &s.kT[u0 * 16 + 16 + l16][0] : s.zpad;
      v16bf bm0 = ld_cont32(bp0);
      v16bf bm1 = ld_cont32(bp1);
      macc[0] = wmma_bf16<false>(a, bm0, macc[0]);
      macc[1] = wmma_bf16<true>(a, bm1, macc[1]);    // reuse_a: same A operand
    }
#pragma unroll
    for (int u2 = 0; u2 < 2; ++u2) {                 // refresh bf16 mirror
      const int u = 2 * wave + u2;
#pragma unroll
      for (int r = 0; r < 8; ++r)
        s.Mbf[half * 8 + r][u * 16 + l16] = (bf16_t)macc[u2][r];
    }
    __syncthreads();                                 // (E)

    // -------- y^T partial on updated M: B = q ------------------------------
    {
      const int g = wave;
      v16bf a = ld_cont32(&s.Mbf[l16][g * 32 + half * 16]);
      const bf16_t* bp = (l16 < DR_B) ? &s.qB[l16][g * 32 + half * 16]
                                      : s.zpad;
      v16bf bm = ld_cont32(bp);
      v8f zc = {0.f, 0.f, 0.f, 0.f, 0.f, 0.f, 0.f, 0.f};
      v8f acc = wmma_bf16<false>(a, bm, zc);
#pragma unroll
      for (int r = 0; r < 8; ++r)
        atomicAdd(&s.redY[half * 8 + r][l16], acc[r]);
    }
    __syncthreads();                                 // (F)

    // -------- wave 0: emit y[:, t, dv-slice], re-zero redY -----------------
    if (wave == 0) {
      float yv[8];
#pragma unroll
      for (int r = 0; r < 8; ++r) {
        yv[r] = s.redY[half * 8 + r][l16];
        s.redY[half * 8 + r][l16] = 0.0f;
      }
      if (l16 < DR_B) {
        const size_t base =
            ((size_t)l16 * DR_S + (size_t)t) * DR_D + dvb + half * 8;
        v4f lo = {yv[0], yv[1], yv[2], yv[3]};
        v4f hi = {yv[4], yv[5], yv[6], yv[7]};
        *(v4f*)(Y + base)     = lo;
        *(v4f*)(Y + base + 4) = hi;
      }
    }
    // next iteration's barrier (A) orders the redY re-zero vs. new atomics
  }
}

extern "C" void kernel_launch(void* const* d_in, const int* in_sizes, int n_in,
                              void* d_out, int out_size, void* d_ws, size_t ws_size,
                              hipStream_t stream) {
  const float* K  = (const float*)d_in[0];
  const float* V  = (const float*)d_in[1];
  const float* Q  = (const float*)d_in[2];
  const float* M0 = (const float*)d_in[3];
  float* Y = (float*)d_out;
  (void)in_sizes; (void)n_in; (void)out_size; (void)d_ws; (void)ws_size;
  delta_rule_scan<<<dim3(DR_D / 16), dim3(256), 0, stream>>>(K, V, Q, M0, Y);
}
